// DMPNNLayer_2954937499917
// MI455X (gfx1250) — matrix operand order repaired and checked by reading, not compiled, not run on hardware
//
#include <hip/hip_runtime.h>
#include <stdint.h>

// ---------------------------------------------------------------------------
// DMPNN layer for MI455X (gfx1250), wave32 + WMMA bf16.
// ---------------------------------------------------------------------------

typedef __attribute__((ext_vector_type(16))) __bf16 v16bf;
typedef __attribute__((ext_vector_type(2)))  __bf16 v2bf;
typedef __attribute__((ext_vector_type(8)))  float  v8f;

#define NODE_DIM 128
#define EDGE_DIM 32
#define HIDDEN   128
#define K1 160   // NODE_DIM + EDGE_DIM
#define K2 128   // HIDDEN
#define KN 256   // NODE_DIM + HIDDEN
#define LN_EPS 1e-5f

union Frag {
    v16bf v;
    uint4 q[2];
};

// Single-instruction ReLU: med3(v, 0, +inf) == clamp to [0, inf)
__device__ __forceinline__ float relu(float v) {
#if __has_builtin(__builtin_amdgcn_fmed3f)
    return __builtin_amdgcn_fmed3f(v, 0.0f, __builtin_inff());
#else
    return v > 0.f ? v : 0.f;
#endif
}

__device__ __forceinline__ uint32_t pack2(float a, float b) {
#if __has_builtin(__builtin_amdgcn_cvt_pk_bf16_f32)
    union { v2bf v; uint32_t u; } c;
    c.v = __builtin_amdgcn_cvt_pk_bf16_f32(a, b);
    return c.u;
#else
    // fptrunc f32->bf16 (RNE); lets the backend use a native cvt if present.
    union { v2bf v; uint32_t u; } c;
    c.v.x = (__bf16)a;
    c.v.y = (__bf16)b;
    return c.u;
#endif
}

__device__ __forceinline__ unsigned short f2bf_raw(float f) {
    return (unsigned short)pack2(f, 0.0f);
}

__device__ __forceinline__ void lds_fence() { __threadfence_block(); }

// A-fragment: 16-bit A 16x32 layout. lane<16 : K = kc*32 + {0..7, 16..23}
//                                    lane>=16: K = kc*32 + {8..15, 24..31}
__device__ __forceinline__ v16bf load_afrag(const char* rowbase, int kbytes) {
    Frag f;
    f.q[0] = *(const uint4*)(rowbase + kbytes);
    f.q[1] = *(const uint4*)(rowbase + kbytes + 32);
    return f.v;
}

// B-fragment from W^T (row = output col n, contiguous in K).
// lane<16 : K = kc*32 + {0..7, 16..23} ; lane>=16 : K = kc*32 + {8..15, 24..31}
__device__ __forceinline__ v16bf load_bfrag(const char* rowbase, int kbytes) {
    Frag f;
    f.q[0] = *(const uint4*)(rowbase + kbytes);
    f.q[1] = *(const uint4*)(rowbase + kbytes + 32);
    return f.v;
}

// ---------------------------------------------------------------------------
// Kernel 1: per-edge MLP + scatter-add into agg
// ---------------------------------------------------------------------------
__global__ void __launch_bounds__(256) edge_kernel(
    const float* __restrict__ x, const long long* __restrict__ ei,
    const float* __restrict__ ea, const float* __restrict__ W1,
    const float* __restrict__ b1, const float* __restrict__ W2,
    const float* __restrict__ b2, float* __restrict__ agg,
    int nTiles, int nEdges)
{
    extern __shared__ char smem[];
    char* w1t   = smem;                 // 128 * 160 * 2 = 40960  (W_e1^T bf16)
    char* w2t   = w1t + 40960;          // 128 * 128 * 2 = 32768  (W_e2^T bf16)
    char* atile = w2t + 32768;          // 8 waves * 16*160*2 = 40960
    char* htile = atile + 40960;        // 8 waves * 16*128*2 = 32768

    const int tid = threadIdx.x;

    // Cooperative load: convert f32 weights -> bf16, transposed (W^T[n][k]).
    {
        unsigned short* w1s = (unsigned short*)w1t;
        for (int e = tid; e < K1 * HIDDEN; e += 256) {
            int k = e >> 7, n = e & 127;
            w1s[n * K1 + k] = f2bf_raw(W1[e]);
        }
        unsigned short* w2s = (unsigned short*)w2t;
        for (int e = tid; e < K2 * HIDDEN; e += 256) {
            int k = e >> 7, n = e & 127;
            w2s[n * K2 + k] = f2bf_raw(W2[e]);
        }
    }
    __syncthreads();

    const int wave = tid >> 5;
    const int lane = tid & 31;
    const int row  = lane & 15;   // A-matrix row / C-matrix column-within-tile
    const int half = lane >> 4;

    char* aw = atile + wave * (16 * K1 * 2);
    char* hw = htile + wave * (16 * K2 * 2);

    for (int tile = blockIdx.x * 8 + wave; tile < nTiles; tile += gridDim.x * 8) {
        const int base = tile * 16;
        const int e    = base + row;

        // ---- gather [src_x || edge_attr] -> bf16 A tile (2 lanes per row) ----
        {
            long long src = ei[(long long)nEdges + e];  // edge_index[1][e]
            const float4* xr = (const float4*)(x + (size_t)src * NODE_DIM);
            uint32_t* arow = (uint32_t*)(aw + row * K1 * 2);
            if (half == 0) {
                #pragma unroll
                for (int t = 0; t < 20; ++t) {          // k = 0..79
                    float4 f = xr[t];
                    arow[t * 2]     = pack2(f.x, f.y);
                    arow[t * 2 + 1] = pack2(f.z, f.w);
                }
            } else {
                #pragma unroll
                for (int t = 20; t < 32; ++t) {         // k = 80..127
                    float4 f = xr[t];
                    arow[t * 2]     = pack2(f.x, f.y);
                    arow[t * 2 + 1] = pack2(f.z, f.w);
                }
                const float4* ar = (const float4*)(ea + (size_t)e * EDGE_DIM);
                #pragma unroll
                for (int t = 0; t < 8; ++t) {           // k = 128..159
                    float4 f = ar[t];
                    arow[64 + t * 2]     = pack2(f.x, f.y);
                    arow[64 + t * 2 + 1] = pack2(f.z, f.w);
                }
            }
        }
        lds_fence();

        // ---- GEMM1: [16 x 160] @ [160 x 128] -> ReLU -> bf16 H in LDS ----
        const int arowbyte = row * K1 * 2;
        for (int jt = 0; jt < 8; ++jt) {
            const int ncol = jt * 16 + row;
            const float bias = b1[ncol];
            v8f acc = {bias, bias, bias, bias, bias, bias, bias, bias};
            const char* brow = w1t + ncol * K1 * 2;
            #pragma unroll
            for (int c = 0; c < 5; ++c) {
                v16bf a = load_afrag(aw + arowbyte, (c * 32 + half * 8) * 2);
                v16bf b = load_bfrag(brow, (c * 32 + half * 8) * 2);
                acc = __builtin_amdgcn_wmma_f32_16x16x32_bf16(
                        false, a, false, b, (short)0, acc, false, false);
            }
            unsigned short* hs = (unsigned short*)hw;
            #pragma unroll
            for (int r = 0; r < 8; r += 2) {
                uint32_t p = pack2(relu(acc[r]), relu(acc[r + 1]));
                int M = r + half * 8;
                hs[M * K2 + ncol]       = (unsigned short)p;         // b16
                hs[(M + 1) * K2 + ncol] = (unsigned short)(p >> 16); // b16 hi
            }
        }
        lds_fence();

        // destination nodes for this wave's C rows (contiguous int64 -> b128s)
        long long dsts[8];
        {
            const longlong2* dp = (const longlong2*)(ei + base + half * 8);
            #pragma unroll
            for (int r = 0; r < 4; ++r) {
                longlong2 d = dp[r];
                dsts[2 * r]     = d.x;
                dsts[2 * r + 1] = d.y;
            }
        }

        // ---- GEMM2: [16 x 128] @ [128 x 128] -> ReLU -> atomic scatter ----
        const int hrowbyte = row * K2 * 2;
        for (int jt = 0; jt < 8; ++jt) {
            const int ncol = jt * 16 + row;
            const float bias = b2[ncol];
            v8f acc = {bias, bias, bias, bias, bias, bias, bias, bias};
            const char* brow = w2t + ncol * K2 * 2;
            #pragma unroll
            for (int c = 0; c < 4; ++c) {
                v16bf a = load_afrag(hw + hrowbyte, (c * 32 + half * 8) * 2);
                v16bf b = load_bfrag(brow, (c * 32 + half * 8) * 2);
                acc = __builtin_amdgcn_wmma_f32_16x16x32_bf16(
                        false, a, false, b, (short)0, acc, false, false);
            }
            #pragma unroll
            for (int r = 0; r < 8; ++r) {
                atomicAdd(agg + (size_t)dsts[r] * HIDDEN + ncol, relu(acc[r]));
            }
        }
        lds_fence();
    }
}

// ---------------------------------------------------------------------------
// Kernel 2: node MLP + residual + LayerNorm
// ---------------------------------------------------------------------------
__global__ void __launch_bounds__(256) node_kernel(
    const float* __restrict__ x, const float* __restrict__ agg,
    const float* __restrict__ Wn, const float* __restrict__ bn,
    const float* __restrict__ gamma, const float* __restrict__ beta,
    float* __restrict__ out, int nTiles)
{
    extern __shared__ char smem[];
    char* wnt = smem;                    // 128 * 256 * 2 = 65536 (W_n^T bf16)
    char* a2  = wnt + 65536;             // 8 waves * 16*256*2 = 65536
    char* rbf = a2 + 65536;              // 8 waves * 16*128*4 = 65536 (f32 rows)

    const int tid = threadIdx.x;
    {
        unsigned short* ws = (unsigned short*)wnt;
        for (int e = tid; e < KN * NODE_DIM; e += 256) {
            int k = e >> 7, n = e & 127;
            ws[n * KN + k] = f2bf_raw(Wn[e]);
        }
    }
    __syncthreads();

    const int wave = tid >> 5;
    const int lane = tid & 31;
    const int row  = lane & 15;
    const int half = lane >> 4;

    char*  aw = a2 + wave * (16 * KN * 2);
    float* Rw = (float*)(rbf + wave * (16 * NODE_DIM * 4));

    for (int tile = blockIdx.x * 8 + wave; tile < nTiles; tile += gridDim.x * 8) {
        const int node = tile * 16 + row;

        // ---- gather [x || agg] -> bf16 A tile ----
        {
            uint32_t* arow = (uint32_t*)(aw + row * KN * 2);
            const float* srcp = (half == 0) ? (x + (size_t)node * NODE_DIM)
                                            : (agg + (size_t)node * HIDDEN);
            const float4* src = (const float4*)srcp;
            #pragma unroll
            for (int t = 0; t < 32; ++t) {
                float4 f = src[t];
                arow[half * 64 + t * 2]     = pack2(f.x, f.y);
                arow[half * 64 + t * 2 + 1] = pack2(f.z, f.w);
            }
        }
        lds_fence();

        // ---- GEMM: [16 x 256] @ [256 x 128] -> ReLU -> f32 LDS ----
        const int arowbyte = row * KN * 2;
        for (int jt = 0; jt < 8; ++jt) {
            const int ncol = jt * 16 + row;
            const float bias = bn[ncol];
            v8f acc = {bias, bias, bias, bias, bias, bias, bias, bias};
            const char* brow = wnt + ncol * KN * 2;
            #pragma unroll
            for (int c = 0; c < 8; ++c) {
                v16bf a = load_afrag(aw + arowbyte, (c * 32 + half * 8) * 2);
                v16bf b = load_bfrag(brow, (c * 32 + half * 8) * 2);
                acc = __builtin_amdgcn_wmma_f32_16x16x32_bf16(
                        false, a, false, b, (short)0, acc, false, false);
            }
            #pragma unroll
            for (int r = 0; r < 8; ++r) {
                int M = r + half * 8;
                Rw[M * NODE_DIM + ncol] = relu(acc[r]);
            }
        }
        lds_fence();

        // ---- residual + LayerNorm: 2 lanes per row, 64 contiguous cols each --
        {
            float s = 0.f, sq = 0.f;
            const float4* rrow = (const float4*)(Rw + row * NODE_DIM + half * 64);
            const float4* xrow = (const float4*)(x + (size_t)node * NODE_DIM + half * 64);
            float4 vals[16];
            #pragma unroll
            for (int t = 0; t < 16; ++t) {
                float4 f = rrow[t];
                float4 xv = xrow[t];
                f.x += xv.x; f.y += xv.y; f.z += xv.z; f.w += xv.w;  // residual
                vals[t] = f;
                s  += f.x + f.y + f.z + f.w;
                sq += f.x * f.x + f.y * f.y + f.z * f.z + f.w * f.w;
            }
            s  += __shfl_xor(s,  16, 32);
            sq += __shfl_xor(sq, 16, 32);
            const float mu   = s * (1.f / 128.f);
            const float var  = sq * (1.f / 128.f) - mu * mu;
            const float rstd = rsqrtf(var + LN_EPS);

            float4* orow = (float4*)(out + (size_t)node * NODE_DIM + half * 64);
            const float4* grow = (const float4*)(gamma + half * 64);
            const float4* brow = (const float4*)(beta + half * 64);
            #pragma unroll
            for (int t = 0; t < 16; ++t) {
                float4 g = grow[t], bb = brow[t], f = vals[t];
                float4 o;
                o.x = (f.x - mu) * rstd * g.x + bb.x;
                o.y = (f.y - mu) * rstd * g.y + bb.y;
                o.z = (f.z - mu) * rstd * g.z + bb.z;
                o.w = (f.w - mu) * rstd * g.w + bb.w;
                orow[t] = o;
            }
        }
        lds_fence();
    }
}

// ---------------------------------------------------------------------------
extern "C" void kernel_launch(void* const* d_in, const int* in_sizes, int n_in,
                              void* d_out, int out_size, void* d_ws, size_t ws_size,
                              hipStream_t stream) {
    const float*      x    = (const float*)d_in[0];
    const long long*  ei   = (const long long*)d_in[1];
    const float*      ea   = (const float*)d_in[2];
    const float*      W1   = (const float*)d_in[3];
    const float*      b1   = (const float*)d_in[4];
    const float*      W2   = (const float*)d_in[5];
    const float*      b2   = (const float*)d_in[6];
    const float*      Wn   = (const float*)d_in[7];
    const float*      bn   = (const float*)d_in[8];
    const float*      gmm  = (const float*)d_in[9];
    const float*      bta  = (const float*)d_in[10];
    float*            out  = (float*)d_out;

    const int nNodes  = in_sizes[0] / NODE_DIM;
    const int nEdges  = in_sizes[1] / 2;
    const int eTiles  = nEdges / 16;      // 40000
    const int nTiles  = nNodes / 16;      // 3125

    float* agg = (float*)d_ws;            // [nNodes, HIDDEN] f32 scratch
    const size_t aggBytes = (size_t)nNodes * HIDDEN * sizeof(float);
    hipMemsetAsync(agg, 0, aggBytes, stream);

    // Edge kernel: 8 waves/block, one 16-edge tile per wave per iteration.
    {
        int blocks = (eTiles + 7) / 8;
        if (blocks > 2500) blocks = 2500;
        size_t shmem = 40960 + 32768 + 40960 + 32768;   // 147456 B
        edge_kernel<<<blocks, 256, shmem, stream>>>(
            x, ei, ea, W1, b1, W2, b2, agg, eTiles, nEdges);
    }

    // Node kernel: 8 waves/block, one 16-node tile per wave per iteration.
    {
        int blocks = (nTiles + 7) / 8;
        size_t shmem = 65536 + 65536 + 65536;           // 196608 B
        node_kernel<<<blocks, 256, shmem, stream>>>(
            x, agg, Wn, bn, gmm, bta, out, nTiles);
    }
}
